// CustomNeRFW_14465449853106
// MI455X (gfx1250) — matrix operand order repaired and checked by reading, not compiled, not run on hardware
//
#include <hip/hip_runtime.h>
#include <hip/hip_bf16.h>
#include <math.h>

// ---------------------------------------------------------------------------
// NeRF-W forward on gfx1250: fp16 WMMA MLP (2 M-tiles per wave to halve
// L2 weight traffic) + per-ray render scan.
// ---------------------------------------------------------------------------

typedef __attribute__((ext_vector_type(16))) _Float16 v16h;
typedef __attribute__((ext_vector_type(8)))  _Float16 v8h;
typedef __attribute__((ext_vector_type(8)))  float    v8f;

#define N_RAYS   2048
#define M_SAMP   64
#define P_TOTAL  (N_RAYS * M_SAMP)     // 131072 points
#define STRIDE   352                   // halves per activation row (max K pad)
#define ROWS     32                    // points per wave (2 WMMA M-tiles)
#define BUF_H    (ROWS * STRIDE)       // 11264 halves per buffer
#define EMB_H    (ROWS * 64)           // 2048 halves

// packed-weight offsets (halves)
#define WO_X0   0
#define WO_X1   16384
#define WO_X2   81920
#define WO_X3   147456
#define WO_X4   212992   // 320x256
#define WO_X5   294912
#define WO_X6   360448
#define WO_X7   425984
#define WO_FIN  491520
#define WO_DIR  557056   // 352x128
#define WO_T0   602112   // 288x128
#define WO_T1   638976
#define WO_T2   655360
#define WO_T3   671744
#define WPACK_TOTAL 688128

struct KArgs {
  const float*    rays_o;
  const float*    rays_d;
  const float*    z;
  const int*      ids;
  const _Float16* wpack;
  const float*    bx[8];
  const float*    b_final;
  const float*    b_dir;
  const float*    tb[4];
  const float*    w_sigma; const float* b_sigma;
  const float*    w_rgb;   const float* b_rgb;
  const float*    w_tsig;  const float* b_tsig;
  const float*    w_trgb;  const float* b_trgb;
  const float*    w_tbeta; const float* b_tbeta;
  const float*    app;
  const float*    tre;
  float* o_ss; float* o_rgbs; float* o_st; float* o_rgbt; float* o_bt;
};

__device__ __forceinline__ float softplusf(float x) {
  return fmaxf(x, 0.f) + log1pf(expf(-fabsf(x)));
}
__device__ __forceinline__ float sigmoidf_(float x) {
  return 1.f / (1.f + expf(-x));
}

// ---------------------------------------------------------------------------
// Weight packing: fp32 (Kreal x N) row-major -> fp16 WMMA B-fragment tiles.
// Dest element idx = ((ntile*nK + ktile)*32 + lane)*16 + e,
// source k = ktile*32 + 8*(lane>>4) + (e&7) + 16*(e>>3) ; n = ntile*16 + (lane&15).
// ---------------------------------------------------------------------------
__global__ void pack_w(const float* __restrict__ src, _Float16* __restrict__ dst,
                       int Kreal, int Kpad, int N) {
  int idx = blockIdx.x * blockDim.x + threadIdx.x;
  int total = Kpad * N;
  if (idx >= total) return;
  int e    = idx & 15;
  int lane = (idx >> 4) & 31;
  int t    = idx >> 9;
  int nK   = Kpad >> 5;
  int kt   = t % nK;
  int nt   = t / nK;
  int k = kt * 32 + ((lane >> 4) << 3) + (e & 7) + ((e >> 3) << 4);
  int n = nt * 16 + (lane & 15);
  float v = (k < Kreal) ? src[k * N + n] : 0.f;
  dst[idx] = (_Float16)v;
}

#define CAT16(lo, hi) \
  __builtin_shufflevector(lo, hi, 0,1,2,3,4,5,6,7,8,9,10,11,12,13,14,15)

// ---------------------------------------------------------------------------
// Per-wave WMMA GEMM over a 32-row tile (2 M-tiles): both A-fragment sets are
// held in VGPRs so each B (weight) fragment load feeds TWO wmmas -> half the
// L2 weight traffic vs a 16-row tile, and paired independent WMMAs for the
// XDL pipeline.
// A layout (16-bit A 16x32): lane&15 = row M; halves 0..7  -> K = kb+8*(l>>4),
//                                      halves 8..15 -> K = kb+16+8*(l>>4).
// D layout (f32 16x16): lane&15 = col N; VGPR v -> row M = v + 8*(l>>4).
// ---------------------------------------------------------------------------
template <int NK, bool RELU>
__device__ __forceinline__ void wave_gemm2(const _Float16* inBuf,
                                           const _Float16* __restrict__ wpack,
                                           int nN, const float* __restrict__ bias,
                                           _Float16* outBuf, int lane) {
  const int p = lane & 15;
  const int h = lane >> 4;
  v16h af0[NK], af1[NK];
  {
    const _Float16* r0 = inBuf + p * STRIDE;
    const _Float16* r1 = inBuf + (16 + p) * STRIDE;
#pragma unroll
    for (int k = 0; k < NK; ++k) {
      v8h lo0 = *(const v8h*)(r0 + k * 32 + h * 8);
      v8h hi0 = *(const v8h*)(r0 + k * 32 + 16 + h * 8);
      af0[k] = CAT16(lo0, hi0);
      v8h lo1 = *(const v8h*)(r1 + k * 32 + h * 8);
      v8h hi1 = *(const v8h*)(r1 + k * 32 + 16 + h * 8);
      af1[k] = CAT16(lo1, hi1);
    }
  }
  for (int n = 0; n < nN; ++n) {
    v8f acc0 = {0.f,0.f,0.f,0.f,0.f,0.f,0.f,0.f};
    v8f acc1 = {0.f,0.f,0.f,0.f,0.f,0.f,0.f,0.f};
    const _Float16* bp = wpack + (size_t)n * NK * 512 + lane * 16;
#pragma unroll
    for (int k = 0; k < NK; ++k) {
      v8h lo = *(const v8h*)(bp + k * 512);
      v8h hi = *(const v8h*)(bp + k * 512 + 8);
      v16h bf = CAT16(lo, hi);
      acc0 = __builtin_amdgcn_wmma_f32_16x16x32_f16(false, af0[k], false, bf,
                                                    (short)0, acc0, false, false);
      acc1 = __builtin_amdgcn_wmma_f32_16x16x32_f16(false, af1[k], false, bf,
                                                    (short)0, acc1, false, false);
    }
    float bv = bias[n * 16 + p];
#pragma unroll
    for (int v = 0; v < 8; ++v) {
      float x0 = acc0[v] + bv;
      float x1 = acc1[v] + bv;
      if (RELU) { x0 = fmaxf(x0, 0.f); x1 = fmaxf(x1, 0.f); }
      outBuf[(v + 8 * h) * STRIDE + n * 16 + p]        = (_Float16)x0;
      outBuf[(16 + v + 8 * h) * STRIDE + n * 16 + p]   = (_Float16)x1;
    }
  }
}

// Small head: one lane per point, full-K serial dot from LDS.
__device__ __forceinline__ float dot_full(const _Float16* row,
                                          const float* __restrict__ w,
                                          int K, int wstride, int col) {
  float s = 0.f;
  for (int k = 0; k < K; ++k)
    s += (float)row[k] * w[k * wstride + col];
  return s;
}

// ---------------------------------------------------------------------------
// Main MLP kernel: 1 wave / block, wave owns a 32-point tile (lane == row).
// ---------------------------------------------------------------------------
__global__ __launch_bounds__(32) void nerf_mlp(KArgs a) {
  __shared__ _Float16 smem[2 * BUF_H + EMB_H];   // 49152 bytes
  _Float16* bufA = smem;
  _Float16* bufB = smem + BUF_H;
  _Float16* E    = smem + 2 * BUF_H;

  const int lane = threadIdx.x;
  const int tile = blockIdx.x;            // 32-point tile
  const int g    = tile * ROWS + lane;    // this lane's point
  const int ray  = g >> 6;
  const int smp  = g & 63;

  const float zv = a.z[ray * 64 + smp];
  const float dx = a.rays_d[ray * 3 + 0];
  const float dy = a.rays_d[ray * 3 + 1];
  const float dz = a.rays_d[ray * 3 + 2];
  const float x0 = a.rays_o[ray * 3 + 0] + dx * zv;
  const float x1 = a.rays_o[ray * 3 + 1] + dy * zv;
  const float x2 = a.rays_o[ray * 3 + 2] + dz * zv;

  // xyz positional embedding: [x,y,z, {sinx,siny,sinz,cosx,cosy,cosz}_j=0..9]
  for (int f = 0; f < 64; ++f) {
    float val;
    if (f < 3) {
      val = (f == 0) ? x0 : (f == 1) ? x1 : x2;
    } else if (f < 63) {
      int t = f - 3, j = t / 6, r = t % 6, ax = r % 3;
      float xv = (ax == 0) ? x0 : (ax == 1) ? x1 : x2;
      float arg = xv * (float)(1 << j);
      val = (r < 3) ? sinf(arg) : cosf(arg);
    } else {
      val = 0.f;
    }
    _Float16 hv = (_Float16)val;
    E[lane * 64 + f] = hv;
    bufA[lane * STRIDE + f] = hv;
  }

  const _Float16* W = a.wpack;
  wave_gemm2<2,  true>(bufA, W + WO_X0, 16, a.bx[0], bufB, lane);
  wave_gemm2<8,  true>(bufB, W + WO_X1, 16, a.bx[1], bufA, lane);
  wave_gemm2<8,  true>(bufA, W + WO_X2, 16, a.bx[2], bufB, lane);
  wave_gemm2<8,  true>(bufB, W + WO_X3, 16, a.bx[3], bufA, lane);

  // skip concat [xyz_emb(63), h(256), pad] -> bufB (320 cols)
  for (int c = 0; c < 320; ++c) {
    _Float16 v;
    if (c < 63)       v = E[lane * 64 + c];
    else if (c < 319) v = bufA[lane * STRIDE + (c - 63)];
    else              v = (_Float16)0.f;
    bufB[lane * STRIDE + c] = v;
  }
  wave_gemm2<10, true>(bufB, W + WO_X4, 16, a.bx[4], bufA, lane);
  wave_gemm2<8,  true>(bufA, W + WO_X5, 16, a.bx[5], bufB, lane);
  wave_gemm2<8,  true>(bufB, W + WO_X6, 16, a.bx[6], bufA, lane);
  wave_gemm2<8,  true>(bufA, W + WO_X7, 16, a.bx[7], bufB, lane);  // h7 -> bufB

  // static sigma head (one point per lane)
  {
    float s = dot_full(bufB + lane * STRIDE, a.w_sigma, 256, 1, 0);
    a.o_ss[g] = softplusf(s + a.b_sigma[0]);
  }

  // feat = h7 @ w_final + b_final (no relu) -> bufA[0..255]
  wave_gemm2<8, false>(bufB, W + WO_FIN, 16, a.b_final, bufA, lane);

  // dir branch extras: cols 256..282 dir_emb, 283..330 app, 331..351 zero
  const int id = a.ids[ray];
  for (int c = 256; c < 352; ++c) {
    int fe = c - 256;
    float val;
    if (fe < 3) {
      val = (fe == 0) ? dx : (fe == 1) ? dy : dz;
    } else if (fe < 27) {
      int t = fe - 3, j = t / 6, r = t % 6, ax = r % 3;
      float dv = (ax == 0) ? dx : (ax == 1) ? dy : dz;
      float arg = dv * (float)(1 << j);
      val = (r < 3) ? sinf(arg) : cosf(arg);
    } else if (fe < 75) {
      val = a.app[id * 48 + (fe - 27)];
    } else {
      val = 0.f;
    }
    bufA[lane * STRIDE + c] = (_Float16)val;
  }
  wave_gemm2<11, true>(bufA, W + WO_DIR, 8, a.b_dir, bufB, lane);  // d -> bufB[0..127]

  // static rgb head
  for (int c = 0; c < 3; ++c) {
    float s = dot_full(bufB + lane * STRIDE, a.w_rgb, 128, 3, c);
    a.o_rgbs[g * 3 + c] = sigmoidf_(s + a.b_rgb[c]);
  }

  // transient branch: cols 0..255 = feat (still in bufA), 256..271 tre, rest 0
  for (int c = 256; c < 288; ++c) {
    int fe = c - 256;
    float val = (fe < 16) ? a.tre[id * 16 + fe] : 0.f;
    bufA[lane * STRIDE + c] = (_Float16)val;
  }
  wave_gemm2<9, true>(bufA, W + WO_T0, 8, a.tb[0], bufB, lane);
  wave_gemm2<4, true>(bufB, W + WO_T1, 8, a.tb[1], bufA, lane);
  wave_gemm2<4, true>(bufA, W + WO_T2, 8, a.tb[2], bufB, lane);
  wave_gemm2<4, true>(bufB, W + WO_T3, 8, a.tb[3], bufA, lane);    // t -> bufA[0..127]

  // transient heads
  {
    const _Float16* trow = bufA + lane * STRIDE;
    float s = dot_full(trow, a.w_tsig, 128, 1, 0);
    a.o_st[g] = softplusf(s + a.b_tsig[0]);
    float b = dot_full(trow, a.w_tbeta, 128, 1, 0);
    a.o_bt[g] = softplusf(b + a.b_tbeta[0]);
    for (int c = 0; c < 3; ++c) {
      float r = dot_full(trow, a.w_trgb, 128, 3, c);
      a.o_rgbt[g * 3 + c] = sigmoidf_(r + a.b_trgb[c]);
    }
  }
}

// ---------------------------------------------------------------------------
// Per-ray volume rendering scan (shared transmittance for both fields).
// ---------------------------------------------------------------------------
__global__ void nerf_render(const float* __restrict__ z,
                            const float* __restrict__ ss,
                            const float* __restrict__ rgbs,
                            const float* __restrict__ st,
                            const float* __restrict__ rgbt,
                            const float* __restrict__ bt,
                            float* __restrict__ out) {
  int r = blockIdx.x * blockDim.x + threadIdx.x;
  if (r >= N_RAYS) return;
  float* o0 = out;            // ray_rgb           (n,3)
  float* o1 = out + 6144;     // static_weights    (n,m)
  float* o2 = out + 137216;   // static_z          (n,1)
  float* o3 = out + 139264;   // transient_sigma   (n,m)
  float* o4 = out + 270336;   // static_ray_rgb    (n,3)
  float* o5 = out + 276480;   // transient_ray_beta(n,1)
  float* o6 = out + 278528;   // transient_ray_rgb (n,3)
  float* o7 = out + 284672;   // transient_weights (n,m)
  float* o8 = out + 415744;   // transient_z       (n,1)

  float T = 1.f;
  float sr0 = 0.f, sr1 = 0.f, sr2 = 0.f;
  float tr0 = 0.f, tr1 = 0.f, tr2 = 0.f;
  float tb = 0.f, sz = 0.f, tz = 0.f;
  for (int i = 0; i < M_SAMP; ++i) {
    int g = r * M_SAMP + i;
    float zv = z[g];
    float delta = (i < M_SAMP - 1) ? (z[g + 1] - zv) : 233.0f;
    float s_s = ss[g], s_t = st[g];
    float as = 1.f - expf(-s_s * delta);
    float at = 1.f - expf(-s_t * delta);
    float ws = T * as;
    float wt = T * at;
    o1[g] = ws;
    o7[g] = wt;
    o3[g] = s_t;
    sr0 += ws * rgbs[g * 3 + 0]; sr1 += ws * rgbs[g * 3 + 1]; sr2 += ws * rgbs[g * 3 + 2];
    tr0 += wt * rgbt[g * 3 + 0]; tr1 += wt * rgbt[g * 3 + 1]; tr2 += wt * rgbt[g * 3 + 2];
    tb += wt * bt[g];
    sz += ws * zv;
    tz += wt * zv;
    T *= expf(-(s_s + s_t) * delta);
  }
  o0[r * 3 + 0] = sr0 + tr0; o0[r * 3 + 1] = sr1 + tr1; o0[r * 3 + 2] = sr2 + tr2;
  o4[r * 3 + 0] = sr0;       o4[r * 3 + 1] = sr1;       o4[r * 3 + 2] = sr2;
  o6[r * 3 + 0] = tr0;       o6[r * 3 + 1] = tr1;       o6[r * 3 + 2] = tr2;
  o2[r] = sz;
  o5[r] = tb + 0.03f;
  o8[r] = tz;
}

// ---------------------------------------------------------------------------
extern "C" void kernel_launch(void* const* d_in, const int* in_sizes, int n_in,
                              void* d_out, int out_size, void* d_ws, size_t ws_size,
                              hipStream_t stream) {
  (void)in_sizes; (void)n_in; (void)out_size; (void)ws_size;

  // workspace layout: per-point floats, then packed fp16 weights
  float* wsf = (float*)d_ws;
  const int P = P_TOTAL;
  float* ws_ss   = wsf;
  float* ws_rgbs = wsf + (size_t)P;
  float* ws_st   = wsf + (size_t)4 * P;
  float* ws_rgbt = wsf + (size_t)5 * P;
  float* ws_bt   = wsf + (size_t)8 * P;
  _Float16* wpack = (_Float16*)(wsf + (size_t)9 * P);

  // --- pack weights (fp32 -> fp16 WMMA B-fragment layout) ---
  struct PackJob { int src_idx; int off; int Kreal, Kpad, N; };
  const PackJob jobs[14] = {
    { 4, WO_X0,  63,  64,  256}, { 5, WO_X1, 256, 256, 256},
    { 6, WO_X2, 256, 256, 256}, { 7, WO_X3, 256, 256, 256},
    { 8, WO_X4, 319, 320, 256}, { 9, WO_X5, 256, 256, 256},
    {10, WO_X6, 256, 256, 256}, {11, WO_X7, 256, 256, 256},
    {22, WO_FIN,256, 256, 256}, {24, WO_DIR,331, 352, 128},
    {28, WO_T0, 272, 288, 128}, {29, WO_T1, 128, 128, 128},
    {30, WO_T2, 128, 128, 128}, {31, WO_T3, 128, 128, 128},
  };
  for (int j = 0; j < 14; ++j) {
    int total = jobs[j].Kpad * jobs[j].N;
    pack_w<<<(total + 255) / 256, 256, 0, stream>>>(
        (const float*)d_in[jobs[j].src_idx], wpack + jobs[j].off,
        jobs[j].Kreal, jobs[j].Kpad, jobs[j].N);
  }

  // --- main MLP ---
  KArgs a;
  a.rays_o = (const float*)d_in[0];
  a.rays_d = (const float*)d_in[1];
  a.z      = (const float*)d_in[2];
  a.ids    = (const int*)d_in[3];
  a.wpack  = wpack;
  for (int i = 0; i < 8; ++i) a.bx[i] = (const float*)d_in[12 + i];
  a.w_sigma = (const float*)d_in[20]; a.b_sigma = (const float*)d_in[21];
  a.b_final = (const float*)d_in[23];
  a.b_dir   = (const float*)d_in[25];
  a.w_rgb   = (const float*)d_in[26]; a.b_rgb   = (const float*)d_in[27];
  for (int i = 0; i < 4; ++i) a.tb[i] = (const float*)d_in[32 + i];
  a.w_tsig  = (const float*)d_in[36]; a.b_tsig  = (const float*)d_in[37];
  a.w_trgb  = (const float*)d_in[38]; a.b_trgb  = (const float*)d_in[39];
  a.w_tbeta = (const float*)d_in[40]; a.b_tbeta = (const float*)d_in[41];
  a.app = (const float*)d_in[42];
  a.tre = (const float*)d_in[43];
  a.o_ss = ws_ss; a.o_rgbs = ws_rgbs; a.o_st = ws_st; a.o_rgbt = ws_rgbt; a.o_bt = ws_bt;

  const int n_tiles = P_TOTAL / ROWS;            // 4096 32-point tiles
  nerf_mlp<<<n_tiles, 32, 0, stream>>>(a);       // 1 wave per block

  // --- render scan ---
  nerf_render<<<(N_RAYS + 255) / 256, 256, 0, stream>>>(
      (const float*)d_in[2], ws_ss, ws_rgbs, ws_st, ws_rgbt, ws_bt, (float*)d_out);
}
